// MultiHeadAttention_64132451664159
// MI455X (gfx1250) — compile-verified, matrix operand order checked by reading, not run on hardware
//
#include <hip/hip_runtime.h>

// ---------------------------------------------------------------------------
// MI455X (gfx1250) multi-head-attention variant:
//   q,k,v = x@W + b ; y = softmax_chunk64(q^T k / 8) ; out = v @ y
// Heavy math: V_WMMA_F32_16X16X32_BF16 (f32 accumulate).
// Data movement: double-buffered LDS, async VMEM->LDS when available.
// ---------------------------------------------------------------------------

typedef __bf16 bf16;
typedef bf16  v16bf __attribute__((ext_vector_type(16)));
typedef float v8f   __attribute__((ext_vector_type(8)));

union FragU { v16bf v; uint4 u[2]; };

#define DIMSZ 1024            // M = N = K = S = dm = 1024 for every GEMM here
#define NBATCH 32

// Async VMEM->LDS path (gfx1250): use if the toolchain exposes the builtins.
#ifndef USE_ASYNC_LDS
#  if defined(__has_builtin)
#    if __has_builtin(__builtin_amdgcn_global_load_async_to_lds_b128) && \
        __has_builtin(__builtin_amdgcn_s_wait_asynccnt)
#      define USE_ASYNC_LDS 1
#    endif
#  endif
#endif
#ifndef USE_ASYNC_LDS
#  define USE_ASYNC_LDS 0
#endif

#if USE_ASYNC_LDS
// Diagnosed param type: '__attribute__((__vector_size__(4*sizeof(int)))) int *'
typedef int v4i __attribute__((vector_size(16)));
typedef v4i* flat_v4i;                                     // flat src pointer
typedef __attribute__((address_space(3))) v4i* lds_v4i;    // LDS dst pointer
#endif

// ---------------------------------------------------------------------------
// Transpose + convert a 1024x1024 fp32 weight to bf16, stored as Wt[n][k].
// ---------------------------------------------------------------------------
__global__ void cvt_wt_kernel(const float* __restrict__ w, bf16* __restrict__ wt) {
    int idx = blockIdx.x * blockDim.x + threadIdx.x;   // 0 .. 1M-1
    int n = idx >> 10;
    int k = idx & 1023;
    wt[idx] = (bf16)w[(size_t)k * DIMSZ + n];
}

// ---------------------------------------------------------------------------
// WMMA GEMM:  C[m,n] = sum_k A[m,k] * Bt[n,k]  (+ bias[n])
//   A : fp32 (AF32=true, converted during staging) or bf16, row-major MxK
//   Bt: bf16, row-major NxK (i.e. B transposed)
//   OUTM: 0 = f32 row-major, 1 = bf16 row-major, 2 = bf16 transposed (C^T[n][m])
// Block: 256 threads = 8 wave32s, tile 128x128, BK=32, double-buffered LDS.
// Wave grid 2(M) x 4(N); each wave owns 4x2 = 8 WMMA 16x16 accumulators.
// ---------------------------------------------------------------------------
template<int OUTM, bool AF32, bool BIAS>
__global__ __launch_bounds__(256)
void gemm_wmma(const void* __restrict__ A, const bf16* __restrict__ Bt,
               const float* __restrict__ bias, void* __restrict__ C,
               size_t strideA, size_t strideBt, size_t strideC) {
    __shared__ __attribute__((aligned(16))) bf16 sA[2][128 * 32];
    __shared__ __attribute__((aligned(16))) bf16 sB[2][128 * 32];

    const int z      = blockIdx.z;
    const int tid    = threadIdx.x;
    const int lane   = tid & 31;
    const int wv     = tid >> 5;
    const int lrow   = lane & 15;     // row/col within 16x16 tile
    const int half   = lane >> 4;     // which K half this lane holds
    const int wm0    = (wv & 1) * 64; // wave offset in M within block tile
    const int wn0    = (wv >> 1) * 32;// wave offset in N within block tile
    const int blockM = blockIdx.y * 128;
    const int blockN = blockIdx.x * 128;

    const float* Af  = (const float*)A + (size_t)z * strideA;   // used if AF32
    const bf16*  Ab  = (const bf16*)A  + (size_t)z * strideA;   // used if !AF32
    const bf16*  Bp  = Bt + (size_t)z * strideBt;
    float*       Cf  = (float*)C + (size_t)z * strideC;
    bf16*        Cb  = (bf16*)C  + (size_t)z * strideC;

    // Stage one 128x32 bf16 tile of A and Bt into LDS buffer `buf`.
    // Each thread owns 2 chunks of 8 bf16 (16 B) per operand.
    auto stage = [&](int kt, int buf) {
        const int k0 = kt * 32;
        #pragma unroll
        for (int r = 0; r < 2; ++r) {
            const int c   = tid + r * 256;   // 512 chunks per tile
            const int row = c >> 2;
            const int cc  = c & 3;
            const size_t gA = (size_t)(blockM + row) * DIMSZ + (k0 + cc * 8);
            const size_t gB = (size_t)(blockN + row) * DIMSZ + (k0 + cc * 8);
            bf16* dA = &sA[buf][row * 32 + cc * 8];
            bf16* dB = &sB[buf][row * 32 + cc * 8];
#if USE_ASYNC_LDS
            __builtin_amdgcn_global_load_async_to_lds_b128(
                (flat_v4i)(Bp + gB), (lds_v4i)dB, 0, 0);
#else
            *(uint4*)dB = *(const uint4*)(Bp + gB);
#endif
            if (AF32) {
                const float* ap = Af + gA;
                float4 f0 = *(const float4*)ap;
                float4 f1 = *(const float4*)(ap + 4);
                bf16 h[8] __attribute__((aligned(16)));
                h[0] = (bf16)f0.x; h[1] = (bf16)f0.y; h[2] = (bf16)f0.z; h[3] = (bf16)f0.w;
                h[4] = (bf16)f1.x; h[5] = (bf16)f1.y; h[6] = (bf16)f1.z; h[7] = (bf16)f1.w;
                *(uint4*)dA = *(const uint4*)h;
            } else {
#if USE_ASYNC_LDS
                __builtin_amdgcn_global_load_async_to_lds_b128(
                    (flat_v4i)(Ab + gA), (lds_v4i)dA, 0, 0);
#else
                *(uint4*)dA = *(const uint4*)(Ab + gA);
#endif
            }
        }
    };

    const v8f vzero = {0.f, 0.f, 0.f, 0.f, 0.f, 0.f, 0.f, 0.f};
    v8f acc[4][2];
    #pragma unroll
    for (int i = 0; i < 4; ++i)
        #pragma unroll
        for (int j = 0; j < 2; ++j) acc[i][j] = vzero;

    const int NT = DIMSZ / 32;
    stage(0, 0);                         // prologue: tile 0 into buffer 0

    for (int kt = 0; kt < NT; ++kt) {
        const int cur = kt & 1;
        const int nxt = cur ^ 1;
        const bool hasNext = (kt + 1 < NT);

        if (hasNext) stage(kt + 1, nxt); // overlap next tile fetch with math

#if USE_ASYNC_LDS
        // Per wave per stage: (AF32 ? 2 : 4) async copies. Async loads retire
        // in order, so allowing the newest stage to stay in flight guarantees
        // the previous stage (buffer `cur`) has landed in LDS.
        if (hasNext) __builtin_amdgcn_s_wait_asynccnt(AF32 ? 2 : 4);
        else         __builtin_amdgcn_s_wait_asynccnt(0);
#endif
        __syncthreads();                 // buf[cur] visible to all waves

        // ---- fragments (ISA 16-bit A layout: 2 x b128 per fragment) ----
        FragU fa[4], fb[2];
        #pragma unroll
        for (int i = 0; i < 4; ++i) {
            const uint4* p =
                (const uint4*)&sA[cur][(wm0 + i * 16 + lrow) * 32 + half * 8];
            fa[i].u[0] = p[0];      // K = half*8 .. half*8+7
            fa[i].u[1] = p[2];      // K = 16+half*8 .. 16+half*8+7
        }
        #pragma unroll
        for (int j = 0; j < 2; ++j) {
            const uint4* p =
                (const uint4*)&sB[cur][(wn0 + j * 16 + lrow) * 32 + half * 8];
            fb[j].u[0] = p[0];
            fb[j].u[1] = p[2];
        }

        // ---- 8 WMMAs per K-step ----
        #pragma unroll
        for (int i = 0; i < 4; ++i)
            #pragma unroll
            for (int j = 0; j < 2; ++j)
                acc[i][j] = __builtin_amdgcn_wmma_f32_16x16x32_bf16(
                    false, fa[i].v, false, fb[j].v, (short)0, acc[i][j],
                    false, false);

        __syncthreads();                 // done reading buf[cur]
    }

    // ---- epilogue: C/D layout = VGPR r holds M=r (lanes 0-15) / M=8+r ----
    #pragma unroll
    for (int i = 0; i < 4; ++i) {
        #pragma unroll
        for (int j = 0; j < 2; ++j) {
            const int mb = blockM + wm0 + i * 16 + half * 8;
            const int nb = blockN + wn0 + j * 16 + lrow;
            const float badd = BIAS ? bias[nb] : 0.f;
            #pragma unroll
            for (int r = 0; r < 8; ++r) {
                const float val = acc[i][j][r] + badd;
                const int m = mb + r;
                if (OUTM == 0)      Cf[(size_t)m * DIMSZ + nb] = val;
                else if (OUTM == 1) Cb[(size_t)m * DIMSZ + nb] = (bf16)val;
                else                Cb[(size_t)nb * DIMSZ + m] = (bf16)val; // C^T
            }
        }
    }
}

// ---------------------------------------------------------------------------
// Chunked softmax: y[b,d,e] -> softmax over 64-wide chunks of e (scaled 1/8),
// output written TRANSPOSED as yt[b,e,d] in bf16 (stage-3 Bt operand layout).
// One thread per (b, d, chunk).
// ---------------------------------------------------------------------------
__global__ void softmax_kernel(const bf16* __restrict__ y, bf16* __restrict__ yt) {
    const int idx = blockIdx.x * blockDim.x + threadIdx.x; // B*1024*16 total
    const int c = idx & 15;
    const int d = (idx >> 4) & 1023;
    const int b = idx >> 14;
    const bf16* p = y + (((size_t)b << 10) + d) * DIMSZ + c * 64;

    const float inv = 0.125f;   // 1 / sqrt(64)
    float v[64];
    float mx = -1e30f;
    #pragma unroll
    for (int e = 0; e < 64; ++e) { v[e] = (float)p[e] * inv; mx = fmaxf(mx, v[e]); }
    float s = 0.f;
    #pragma unroll
    for (int e = 0; e < 64; ++e) { v[e] = __expf(v[e] - mx); s += v[e]; }
    const float r = 1.f / s;

    bf16* q = yt + ((size_t)b << 20) + d;
    #pragma unroll
    for (int e = 0; e < 64; ++e)
        q[(size_t)(c * 64 + e) << 10] = (bf16)(v[e] * r);
}

// ---------------------------------------------------------------------------
// Launch: inputs (x, Wq, bq, Wk, bk, Wv, bv) fp32; out fp32 [32,1024,1024].
// ---------------------------------------------------------------------------
extern "C" void kernel_launch(void* const* d_in, const int* in_sizes, int n_in,
                              void* d_out, int out_size, void* d_ws, size_t ws_size,
                              hipStream_t stream) {
    const float* x  = (const float*)d_in[0];
    const float* Wq = (const float*)d_in[1];
    const float* bq = (const float*)d_in[2];
    const float* Wk = (const float*)d_in[3];
    const float* bk = (const float*)d_in[4];
    const float* Wv = (const float*)d_in[5];
    const float* bv = (const float*)d_in[6];
    float* out = (float*)d_out;

    const size_t MM = (size_t)DIMSZ * DIMSZ;   // 1M elems (per-batch matrix)

    // Workspace carve-up (bf16 elements): 3*1M + 5*32M = ~326 MB
    bf16* wqt  = (bf16*)d_ws;
    bf16* wkt  = wqt  + MM;
    bf16* wvt  = wkt  + MM;
    bf16* qt   = wvt  + MM;            // Q^T  [b][dm][S]
    bf16* kt   = qt   + NBATCH * MM;   // K^T  [b][dm][S]
    bf16* vb   = kt   + NBATCH * MM;   // V    [b][S][dm]
    bf16* ybf  = vb   + NBATCH * MM;   // scores [b][d][e]
    bf16* ytbf = ybf  + NBATCH * MM;   // softmax probs, transposed [b][e][d]

    // 1) weights -> bf16, transposed
    cvt_wt_kernel<<<MM / 256, 256, 0, stream>>>(Wq, wqt);
    cvt_wt_kernel<<<MM / 256, 256, 0, stream>>>(Wk, wkt);
    cvt_wt_kernel<<<MM / 256, 256, 0, stream>>>(Wv, wvt);

    dim3 grid(DIMSZ / 128, DIMSZ / 128, NBATCH);

    // 2) projections (fp32 A converted in staging; bias added)
    //    Q,K stored transposed for stage 3/4 operand layouts; V stored natural.
    gemm_wmma<2, true, true><<<grid, 256, 0, stream>>>(x, wqt, bq, qt, MM, 0, MM);
    gemm_wmma<2, true, true><<<grid, 256, 0, stream>>>(x, wkt, bk, kt, MM, 0, MM);
    gemm_wmma<1, true, true><<<grid, 256, 0, stream>>>(x, wvt, bv, vb, MM, 0, MM);

    // 3) scores: y[d,e] = sum_s Qt[d,s] * Kt[e,s]  (A=Qt, Bt=Kt)
    gemm_wmma<1, false, false><<<grid, 256, 0, stream>>>(qt, kt, nullptr, ybf, MM, MM, MM);

    // 4) chunked softmax, output transposed bf16
    softmax_kernel<<<(NBATCH * DIMSZ * 16) / 256, 256, 0, stream>>>(ybf, ytbf);

    // 5) out[s,e] = sum_d V[s,d] * Yt[e,d]  (A=V, Bt=Yt), fp32 output
    gemm_wmma<0, false, false><<<grid, 256, 0, stream>>>(vb, ytbf, nullptr, out, MM, MM, MM);
}
